// KFCContrastiveLossV2_11605001634113
// MI455X (gfx1250) — matrix operand order repaired and checked by reading, not compiled, not run on hardware
//
#include <hip/hip_runtime.h>
#include <hip/hip_bf16.h>

typedef _Float16 v16h __attribute__((ext_vector_type(16)));
typedef _Float16 v8h  __attribute__((ext_vector_type(8)));
typedef float    v8f  __attribute__((ext_vector_type(8)));

#define NROWS    8192
#define BHALF    4096
#define DDIM     512
#define INV_BETA 12.5f
#define NSPLIT   8
#define AK       520   // padded LDS row stride in halves: 1040B = 260 dwords, %64==4 -> conflict-free b128 gathers

// ---------------------------------------------------------------------------
// Kernel 1: row normalization + f16 hi/lo split.
// x = concat(x1, x2); xn = x / max(||x||, 1e-8); xh = f16(xn); xl = f16(xn - xh)
// ---------------------------------------------------------------------------
__global__ __launch_bounds__(256) void k_normalize(const float* __restrict__ x1,
                                                   const float* __restrict__ x2,
                                                   float* __restrict__ xn,
                                                   _Float16* __restrict__ xh,
                                                   _Float16* __restrict__ xl)
{
    const int r = blockIdx.x;
    const int t = threadIdx.x;
    const float* src = (r < BHALF) ? (x1 + (size_t)r * DDIM)
                                   : (x2 + (size_t)(r - BHALF) * DDIM);
    float v0 = src[t];
    float v1 = src[t + 256];
    __shared__ float red[256];
    red[t] = v0 * v0 + v1 * v1;
    __syncthreads();
    for (int o = 128; o > 0; o >>= 1) {
        if (t < o) red[t] += red[t + o];
        __syncthreads();
    }
    const float inv = 1.0f / fmaxf(sqrtf(red[0]), 1e-8f);
    const float a0 = v0 * inv, a1 = v1 * inv;
    const size_t base = (size_t)r * DDIM;
    xn[base + t]       = a0;
    xn[base + t + 256] = a1;
    const _Float16 h0 = (_Float16)a0, h1 = (_Float16)a1;
    xh[base + t]       = h0;
    xh[base + t + 256] = h1;
    xl[base + t]       = (_Float16)(a0 - (float)h0);
    xl[base + t + 256] = (_Float16)(a1 - (float)h1);
}

// ---------------------------------------------------------------------------
// Kernel 2: d[i] = dot(x1n_i, x2n_i) in full f32.
// ---------------------------------------------------------------------------
__global__ __launch_bounds__(256) void k_dvec(const float* __restrict__ xn,
                                              float* __restrict__ dvec)
{
    const int r = blockIdx.x;
    const int t = threadIdx.x;
    const float* a = xn + (size_t)r * DDIM;
    const float* b = xn + (size_t)(r + BHALF) * DDIM;
    __shared__ float red[256];
    red[t] = a[t] * b[t] + a[t + 256] * b[t + 256];
    __syncthreads();
    for (int o = 128; o > 0; o >>= 1) {
        if (t < o) red[t] += red[t + o];
        __syncthreads();
    }
    if (t == 0) dvec[r] = red[0];
}

// ---------------------------------------------------------------------------
// Kernel 3: bias_margin[i] = mean over M=64 of bias_map row i.
// ---------------------------------------------------------------------------
__global__ __launch_bounds__(256) void k_bias(const float* __restrict__ bias_map,
                                              float* __restrict__ bm)
{
    const int i = blockIdx.x * 256 + threadIdx.x;
    const float4* p = (const float4*)(bias_map + (size_t)i * 64);
    float s = 0.f;
#pragma unroll
    for (int t = 0; t < 16; ++t) {
        const float4 q = p[t];
        s += q.x + q.y + q.z + q.w;
    }
    bm[i] = s * (1.0f / 64.0f);
}

// ---------------------------------------------------------------------------
// Kernel 4: fused Gram-matrix/exp/row-sum with f16 hi/lo split WMMA.
// Block = 256 threads (8 waves); each wave owns 16 rows; block owns 128 rows.
// grid.y splits the 512 column tiles 8 ways (512 blocks total).
// LDS: A panel (128 rows x 512 K, hi+lo) + one B panel (16 cols x 512 K, hi+lo)
//      = 2*128*520*2 + 2*16*520*2 = 299,520 B of the WGP's 320 KB.
// The NEXT B panel is register-double-buffered: each thread holds its 128 B
// slice (4+4 uint4) so the global loads for tile j+1 are in flight while the
// WMMAs for tile j execute; their s_wait lands at the LDS store after compute.
// ---------------------------------------------------------------------------
__global__ __launch_bounds__(256) void k_rowsum(const _Float16* __restrict__ xh,
                                                const _Float16* __restrict__ xl,
                                                float* __restrict__ Spart)
{
    __shared__ alignas(16) _Float16 Ah[128 * AK];
    __shared__ alignas(16) _Float16 Al[128 * AK];
    __shared__ alignas(16) _Float16 Bh[16 * AK];
    __shared__ alignas(16) _Float16 Bl[16 * AK];

    const int tid     = threadIdx.x;
    const int rowBase = blockIdx.x * 128;

    // Stage A panel: 128 rows x 512 halves (hi and lo), b128 chunks, coalesced.
    for (int c = tid; c < 128 * (DDIM / 8); c += 256) {
        const int r  = c >> 6;
        const int k8 = (c & 63) << 3;
        const size_t g = ((size_t)(rowBase + r) << 9) + (size_t)k8;
        *(uint4*)(&Ah[r * AK + k8]) = *(const uint4*)(xh + g);
        *(uint4*)(&Al[r * AK + k8]) = *(const uint4*)(xl + g);
    }

    const int lane = tid & 31;
    const int wave = tid >> 5;
    const int m    = lane & 15;   // A row-in-tile == B col-in-tile for this lane
    const int hk   = lane >> 4;   // K-half selector per ISA 16-bit layouts
    const int aOff = (wave * 16 + m) * AK;
    const int bOff = m * AK;

    // Register prefetch buffers for the B panel (4 chunks hi + 4 chunks lo).
    uint4 pbh[4], pbl[4];
    auto loadB = [&](int jt) {
#pragma unroll
        for (int q = 0; q < 4; ++q) {
            const int c  = tid + q * 256;          // 0..1023
            const int r  = c >> 6;
            const int k8 = (c & 63) << 3;
            const size_t g = ((size_t)(jt * 16 + r) << 9) + (size_t)k8;
            pbh[q] = *(const uint4*)(xh + g);
            pbl[q] = *(const uint4*)(xl + g);
        }
    };
    auto storeB = [&]() {
#pragma unroll
        for (int q = 0; q < 4; ++q) {
            const int c  = tid + q * 256;
            const int r  = c >> 6;
            const int k8 = (c & 63) << 3;
            *(uint4*)(&Bh[r * AK + k8]) = pbh[q];
            *(uint4*)(&Bl[r * AK + k8]) = pbl[q];
        }
    };

    constexpr int NT = 512 / NSPLIT;   // column tiles per block
    const int jt0 = blockIdx.y * NT;
    loadB(jt0);                        // first B panel in flight alongside A staging

    float psum[8] = {0.f, 0.f, 0.f, 0.f, 0.f, 0.f, 0.f, 0.f};

    for (int j = 0; j < NT; ++j) {
        const int jt = jt0 + j;
        storeB();                      // waits on prefetch loads here (post-compute)
        __syncthreads();               // B panel visible to all waves
        if (j + 1 < NT) loadB(jt + 1); // prefetch next panel (uniform branch)

        v8f acc{};
#pragma unroll 4
        for (int kc = 0; kc < 16; ++kc) {
            const int kOff = kc * 32;
            // A 16x32 f16 operand: lanes 0-15 -> K {0..7,16..23}, lanes 16-31 -> +8
            const v8h a0h = *(const v8h*)(&Ah[aOff + kOff + hk * 8]);
            const v8h a1h = *(const v8h*)(&Ah[aOff + kOff + 16 + hk * 8]);
            const v8h a0l = *(const v8h*)(&Al[aOff + kOff + hk * 8]);
            const v8h a1l = *(const v8h*)(&Al[aOff + kOff + 16 + hk * 8]);
            // B 32x16 f16 operand: lanes 0-15 -> K {0..7,16..23}, lanes 16-31 -> +8
            const v8h b0h = *(const v8h*)(&Bh[bOff + kOff + hk * 8]);
            const v8h b1h = *(const v8h*)(&Bh[bOff + kOff + 16 + hk * 8]);
            const v8h b0l = *(const v8h*)(&Bl[bOff + kOff + hk * 8]);
            const v8h b1l = *(const v8h*)(&Bl[bOff + kOff + 16 + hk * 8]);

            const v16h Avh = __builtin_shufflevector(a0h, a1h, 0,1,2,3,4,5,6,7,8,9,10,11,12,13,14,15);
            const v16h Avl = __builtin_shufflevector(a0l, a1l, 0,1,2,3,4,5,6,7,8,9,10,11,12,13,14,15);
            const v16h Bvh = __builtin_shufflevector(b0h, b1h, 0,1,2,3,4,5,6,7,8,9,10,11,12,13,14,15);
            const v16h Bvl = __builtin_shufflevector(b0l, b1l, 0,1,2,3,4,5,6,7,8,9,10,11,12,13,14,15);

            // hi*hi + hi*lo + lo*hi  (lo*lo ~ 2^-22, negligible)
            acc = __builtin_amdgcn_wmma_f32_16x16x32_f16(false, Avh, false, Bvh,
                                                         (short)0, acc, false, false);
            acc = __builtin_amdgcn_wmma_f32_16x16x32_f16(false, Avh, false, Bvl,
                                                         (short)0, acc, false, false);
            acc = __builtin_amdgcn_wmma_f32_16x16x32_f16(false, Avl, false, Bvh,
                                                         (short)0, acc, false, false);
        }

        // exp((xn_i . xn_j)/beta), zeroing the masked diagonal, into row partials.
        // C/D layout: reg v, lanes 0-15 -> (row v, col lane); lanes 16-31 -> (row v+8, col lane-16)
        const bool diag = (rowBase + wave * 16) == (jt * 16);
#pragma unroll
        for (int v = 0; v < 8; ++v) {
            float e = __expf(acc[v] * INV_BETA);
            if (diag && (lane == v || lane == v + 24)) e = 0.f;
            psum[v] += e;
        }
        __syncthreads();               // all waves done reading B before overwrite
    }

    // Reduce each register's partial within each 16-lane half -> 16 row sums.
    float* Sout = Spart + (size_t)blockIdx.y * NROWS + rowBase + wave * 16;
#pragma unroll
    for (int v = 0; v < 8; ++v) {
        float p = psum[v];
#pragma unroll
        for (int o = 8; o > 0; o >>= 1) p += __shfl_xor(p, o, 16);
        if (lane == 0)  Sout[v]     = p;
        if (lane == 16) Sout[8 + v] = p;
    }
}

// ---------------------------------------------------------------------------
// Kernel 5: loss = -mean(log(num/den)), deterministic fixed-tree reduction.
// ---------------------------------------------------------------------------
__global__ __launch_bounds__(256) void k_loss(const float* __restrict__ Spart,
                                              const float* __restrict__ dvec,
                                              const float* __restrict__ bm,
                                              float* __restrict__ out)
{
    __shared__ float red[256];
    const int t = threadIdx.x;
    float acc = 0.f;
    for (int i = t; i < NROWS; i += 256) {
        float s = 0.f;
#pragma unroll
        for (int p = 0; p < NSPLIT; ++p) s += Spart[p * NROWS + i];
        const float d2  = dvec[i & (BHALF - 1)];
        const float b   = bm[i];
        const float num = __expf((d2 - b) * INV_BETA);
        const float den = s - __expf(d2 * INV_BETA) + num;
        acc += (d2 - b) * INV_BETA - __logf(den);
    }
    red[t] = acc;
    __syncthreads();
    for (int o = 128; o > 0; o >>= 1) {
        if (t < o) red[t] += red[t + o];
        __syncthreads();
    }
    if (t == 0) out[0] = -red[0] / (float)NROWS;
}

// ---------------------------------------------------------------------------
// Kernel 6: per-race mean of bias_margin (4 segments), deterministic.
// ---------------------------------------------------------------------------
__global__ __launch_bounds__(256) void k_margins(const int* __restrict__ races,
                                                 const float* __restrict__ bm,
                                                 float* __restrict__ out)
{
    const int t = threadIdx.x;
    float tot[4] = {0.f, 0.f, 0.f, 0.f};
    float cnt[4] = {0.f, 0.f, 0.f, 0.f};
    for (int i = t; i < NROWS; i += 256) {
        const int r = races[i & (BHALF - 1)] & 3;
        tot[r] += bm[i];
        cnt[r] += 1.f;
    }
    __shared__ float st[4][256];
    __shared__ float sc[4][256];
#pragma unroll
    for (int s = 0; s < 4; ++s) { st[s][t] = tot[s]; sc[s][t] = cnt[s]; }
    __syncthreads();
    for (int o = 128; o > 0; o >>= 1) {
        if (t < o) {
#pragma unroll
            for (int s = 0; s < 4; ++s) {
                st[s][t] += st[s][t + o];
                sc[s][t] += sc[s][t + o];
            }
        }
        __syncthreads();
    }
    if (t < 4) out[1 + t] = (sc[t][0] > 0.f) ? (st[t][0] / sc[t][0]) : 0.f;
}

// ---------------------------------------------------------------------------
// Launcher. Workspace layout (bytes):
//   xn  f32  [8192*512]          @ 0          (16,777,216)
//   xh  f16  [8192*512]          @ 16,777,216 ( 8,388,608)
//   xl  f16  [8192*512]          @ 25,165,824 ( 8,388,608)
//   Spart f32 [NSPLIT*8192]      @ 33,554,432 (   262,144)
//   bm  f32  [8192]              @ 33,816,576 (    32,768)
//   dvec f32 [4096]              @ 33,849,344 (    16,384)
//   total ~33.9 MB
// ---------------------------------------------------------------------------
extern "C" void kernel_launch(void* const* d_in, const int* in_sizes, int n_in,
                              void* d_out, int out_size, void* d_ws, size_t ws_size,
                              hipStream_t stream)
{
    (void)in_sizes; (void)n_in; (void)out_size; (void)ws_size;
    const float* x1       = (const float*)d_in[0];
    const float* x2       = (const float*)d_in[1];
    const int*   races    = (const int*)d_in[2];
    const float* bias_map = (const float*)d_in[3];
    float* out = (float*)d_out;

    char* ws = (char*)d_ws;
    float*    xn    = (float*)(ws);
    _Float16* xh    = (_Float16*)(ws + 16777216);
    _Float16* xl    = (_Float16*)(ws + 25165824);
    float*    Spart = (float*)(ws + 33554432);
    float*    bm    = (float*)(ws + 33816576);
    float*    dvec  = (float*)(ws + 33849344);

    k_normalize<<<NROWS, 256, 0, stream>>>(x1, x2, xn, xh, xl);
    k_dvec<<<BHALF, 256, 0, stream>>>(xn, dvec);
    k_bias<<<NROWS / 256, 256, 0, stream>>>(bias_map, bm);
    k_rowsum<<<dim3(NROWS / 128, NSPLIT), 256, 0, stream>>>(xh, xl, Spart);
    k_loss<<<1, 256, 0, stream>>>(Spart, dvec, bm, out);
    k_margins<<<1, 256, 0, stream>>>(races, bm, out);
}
